// ISTAConv_31318901523050
// MI455X (gfx1250) — matrix operand order repaired and checked by reading, not compiled, not run on hardware
//
#include <hip/hip_runtime.h>

typedef __attribute__((ext_vector_type(16))) _Float16 v16h;
typedef __attribute__((ext_vector_type(8)))  _Float16 v8h;
typedef __attribute__((ext_vector_type(8)))  float    v8f;

#define Mdim 8192
#define Ndim 16384
#define Ecnt 2097152
#define Bdim 8
#define IMGD 128

// ---------------- elementwise helpers ----------------
__global__ void copy_f32(float* __restrict__ dst, const float* __restrict__ src, int n) {
    int i = blockIdx.x * blockDim.x + threadIdx.x;
    if (i < n) dst[i] = src[i];
}

__global__ void zero_f32(float* __restrict__ dst, int n) {
    int i = blockIdx.x * blockDim.x + threadIdx.x;
    if (i < n) dst[i] = 0.0f;
}

// resid[b,m] -= vals[e] * z[b,col]*u[b,col]   (resid pre-initialized to y)
__global__ void scatter_xAT(const float* __restrict__ vals, const int* __restrict__ rows,
                            const int* __restrict__ cols, const float* __restrict__ z,
                            const float* __restrict__ u, float* __restrict__ resid) {
    int e = blockIdx.x * blockDim.x + threadIdx.x;
    if (e >= Ecnt) return;
    if (e + 8192 < Ecnt) {   // stream-ahead prefetch of COO arrays (global_prefetch_b8)
        __builtin_prefetch(&vals[e + 8192], 0, 3);
        __builtin_prefetch(&rows[e + 8192], 0, 3);
        __builtin_prefetch(&cols[e + 8192], 0, 3);
    }
    float va = vals[e];
    int r = rows[e], c = cols[e];
#pragma unroll
    for (int b = 0; b < Bdim; ++b) {
        float x = z[b * Ndim + c] * u[b * Ndim + c];
        atomicAdd(&resid[b * Mdim + r], -va * x);
    }
}

// v[b,n] += vals[e] * resid[b,row]
__global__ void scatter_xA(const float* __restrict__ vals, const int* __restrict__ rows,
                           const int* __restrict__ cols, const float* __restrict__ resid,
                           float* __restrict__ v) {
    int e = blockIdx.x * blockDim.x + threadIdx.x;
    if (e >= Ecnt) return;
    if (e + 8192 < Ecnt) {
        __builtin_prefetch(&vals[e + 8192], 0, 3);
        __builtin_prefetch(&rows[e + 8192], 0, 3);
        __builtin_prefetch(&cols[e + 8192], 0, 3);
    }
    float va = vals[e];
    int r = rows[e], c = cols[e];
#pragma unroll
    for (int b = 0; b < Bdim; ++b)
        atomicAdd(&v[b * Ndim + c], va * resid[b * Mdim + r]);
}

// v = u * v ; norm[b] += sum(v^2) (block-level reduction, one atomic per block)
__global__ void vnorm_kernel(const float* __restrict__ u, float* __restrict__ v,
                             float* __restrict__ norm) {
    __shared__ float red[256];
    int b = blockIdx.y;
    int i = b * Ndim + blockIdx.x * 256 + threadIdx.x;
    float val = u[i] * v[i];
    v[i] = val;
    red[threadIdx.x] = val * val;
    __syncthreads();
    for (int s = 128; s > 0; s >>= 1) {
        if (threadIdx.x < s) red[threadIdx.x] += red[threadIdx.x + s];
        __syncthreads();
    }
    if (threadIdx.x == 0) atomicAdd(&norm[b], red[0]);
}

// r = z - eta * v / max(1, ||v||/BCLIP)   (BCLIP = 1)
__global__ void rstep_kernel(const float* __restrict__ z, const float* __restrict__ v,
                             const float* __restrict__ norm, const float* __restrict__ eta,
                             float* __restrict__ r) {
    int i = blockIdx.x * 256 + threadIdx.x;
    int b = i >> 14;
    float scale = fmaxf(1.0f, sqrtf(norm[b]));
    r[i] = z[i] - eta[0] * v[i] / scale;
}

// ---------------- conv1: 1 -> 32 channels, K=9 zero-padded into K=32 WMMA ----------------
__global__ void conv1_wmma(const float* __restrict__ r, const float* __restrict__ W1,
                           _Float16* __restrict__ act1) {
    int wave = (blockIdx.x << 3) + (threadIdx.x >> 5);  // 16-pixel tile id
    int lane = threadIdx.x & 31;
    int hi = lane >> 4;
    int n = lane & 15;

    // A tile: 16 pixels x 32 K  (only K<9 used: the 3x3 taps of the single input channel)
    int p = (wave << 4) + n;            // A-row = lane&15 -> pixel
    int b = p >> 14, idx = p & 16383, py = idx >> 7, px = idx & 127;
    v16h a = {};
#pragma unroll
    for (int e = 0; e < 16; ++e) {
        int K = (e < 8 ? e : e + 8) + hi * 8;
        _Float16 val = (_Float16)0.0f;
        if (K < 9) {
            int yy = py + K / 3 - 1, xx = px + K % 3 - 1;
            if (yy >= 0 && yy < IMGD && xx >= 0 && xx < IMGD)
                val = (_Float16)r[(b << 14) + (yy << 7) + xx];
        }
        a[e] = val;
    }
    // B tiles: K x 16 ocs, two oc groups. W1 layout (3,3,1,32) -> [tap*32 + oc]
    v16h b0 = {}, b1 = {};
#pragma unroll
    for (int e = 0; e < 16; ++e) {
        int K = (e < 8 ? e : e + 8) + hi * 8;
        if (K < 9) {
            b0[e] = (_Float16)W1[K * 32 + n];
            b1[e] = (_Float16)W1[K * 32 + 16 + n];
        }
    }
    v8f c0 = {}, c1 = {};
    c0 = __builtin_amdgcn_wmma_f32_16x16x32_f16(false, a, false, b0, (short)0, c0, false, false);
    c1 = __builtin_amdgcn_wmma_f32_16x16x32_f16(false, a, false, b1, (short)0, c1, false, false);
#pragma unroll
    for (int rr = 0; rr < 8; ++rr) {
        int pm = (wave << 4) + rr + hi * 8;
        act1[pm * 32 + n]      = (_Float16)fmaxf(c0[rr], 0.0f);
        act1[pm * 32 + 16 + n] = (_Float16)fmaxf(c1[rr], 0.0f);
    }
}

// ---- conv2: 32 -> 32 channels, 9 taps x (K=32) WMMA ----
// A: two global_load_b128 per tap (contiguous 8xf16 runs per lane).
// B: W2 transposed in LDS to [tap][oc][ic] so a lane's 16 K-values are one
//    contiguous 32B run -> ds_load_b128 pairs.
__global__ void conv2_wmma(const _Float16* __restrict__ act1, const float* __restrict__ W2,
                           _Float16* __restrict__ act2) {
    __shared__ _Float16 w2t[9 * 32 * 32];   // [t][oc][ic], 18 KB of 320 KB/WGP
    for (int i = threadIdx.x; i < 9 * 32 * 32; i += 256) {
        int oc = i & 31, ic = (i >> 5) & 31, t = i >> 10;   // src layout [t][ic][oc]
        w2t[t * 1024 + oc * 32 + ic] = (_Float16)W2[i];
    }
    __syncthreads();

    int wave = (blockIdx.x << 3) + (threadIdx.x >> 5);
    int lane = threadIdx.x & 31;
    int hi = lane >> 4;
    int n = lane & 15;
    int p = (wave << 4) + n;
    int b = p >> 14, idx = p & 16383, py = idx >> 7, px = idx & 127;

    v8f c0 = {}, c1 = {};
#pragma unroll
    for (int t = 0; t < 9; ++t) {
        int yy = py + t / 3 - 1, xx = px + t % 3 - 1;
        bool inb = (yy >= 0 && yy < IMGD && xx >= 0 && xx < IMGD);
        int np = (b << 14) + (yy << 7) + xx;
        v16h a = {};
        if (inb) {
            const v8h* pa = (const v8h*)(act1 + (size_t)np * 32);  // 64B-aligned row
            v8h lo = pa[2 * hi];    // ic = hi*16 .. hi*16+7
            v8h hh = pa[2 * hi + 1]; // ic = hi*16+8 .. hi*16+15
#pragma unroll
            for (int e = 0; e < 8; ++e) { a[e] = lo[e]; a[e + 8] = hh[e]; }
        }
        v16h b0 = *(const v16h*)&w2t[t * 1024 + n * 32 + hi * 16];         // oc = n
        v16h b1 = *(const v16h*)&w2t[t * 1024 + (16 + n) * 32 + hi * 16];  // oc = 16+n
        c0 = __builtin_amdgcn_wmma_f32_16x16x32_f16(false, a, false, b0, (short)0, c0, false, false);
        c1 = __builtin_amdgcn_wmma_f32_16x16x32_f16(false, a, false, b1, (short)0, c1, false, false);
    }
#pragma unroll
    for (int rr = 0; rr < 8; ++rr) {
        int pm = (wave << 4) + rr + hi * 8;
        act2[pm * 32 + n]      = (_Float16)fmaxf(c0[rr], 0.0f);
        act2[pm * 32 + 16 + n] = (_Float16)fmaxf(c1[rr], 0.0f);
    }
}

// ---------------- conv3 (32 -> 1, VALU, b128-vectorized) fused with z = relu(r + x) --------
__global__ void conv3_zupdate(const _Float16* __restrict__ act2, const float* __restrict__ W3,
                              const float* __restrict__ r, float* __restrict__ zout) {
    __shared__ float w3s[288];
    for (int i = threadIdx.x; i < 288; i += 256) w3s[i] = W3[i];
    __syncthreads();
    int p = blockIdx.x * 256 + threadIdx.x;
    int b = p >> 14, idx = p & 16383, py = idx >> 7, px = idx & 127;
    float acc = 0.0f;
#pragma unroll
    for (int t = 0; t < 9; ++t) {
        int yy = py + t / 3 - 1, xx = px + t % 3 - 1;
        if (yy < 0 || yy >= IMGD || xx < 0 || xx >= IMGD) continue;
        const v8h* pa = (const v8h*)(act2 + ((size_t)((b << 14) + (yy << 7) + xx)) * 32);
#pragma unroll
        for (int q = 0; q < 4; ++q) {
            v8h chunk = pa[q];
#pragma unroll
            for (int e = 0; e < 8; ++e)
                acc += (float)chunk[e] * w3s[t * 32 + q * 8 + e];
        }
    }
    zout[p] = fmaxf(r[p] + acc, 0.0f);
}

extern "C" void kernel_launch(void* const* d_in, const int* in_sizes, int n_in,
                              void* d_out, int out_size, void* d_ws, size_t ws_size,
                              hipStream_t stream) {
    (void)in_sizes; (void)n_in; (void)out_size; (void)ws_size;
    const float* A_vals = (const float*)d_in[0];
    const int*   A_rows = (const int*)d_in[1];
    const int*   A_cols = (const int*)d_in[2];
    const float* z0     = (const float*)d_in[3];
    const float* u      = (const float*)d_in[4];
    const float* y      = (const float*)d_in[5];
    const float* W1     = (const float*)d_in[6];
    const float* W2     = (const float*)d_in[7];
    const float* W3     = (const float*)d_in[8];
    const float* etas   = (const float*)d_in[9];
    float* zout = (float*)d_out;

    // workspace carve-up (~18 MB total; whole working set is L2-resident on MI455X)
    float* resid = (float*)d_ws;                       // B*M   = 65536 f32
    float* v     = resid + Bdim * Mdim;                // B*N   = 131072 f32
    float* r     = v + Bdim * Ndim;                    // B*N   = 131072 f32
    float* norm  = r + Bdim * Ndim;                    // 8 f32 (padded to 64 -> keeps act1 64B-aligned)
    _Float16* act1 = (_Float16*)(norm + 64);           // B*16384*32 f16 = 8 MB
    _Float16* act2 = act1 + (size_t)Bdim * 16384 * 32; // 8 MB

    for (int j = 0; j < 3; ++j) {
        const float* zin = (j == 0) ? z0 : zout;

        // resid = y - (z*u) @ A^T
        copy_f32<<<256, 256, 0, stream>>>(resid, y, Bdim * Mdim);
        scatter_xAT<<<Ecnt / 256, 256, 0, stream>>>(A_vals, A_rows, A_cols, zin, u, resid);

        // v = u * (resid @ A); per-sample norm
        zero_f32<<<512, 256, 0, stream>>>(v, Bdim * Ndim);
        zero_f32<<<1, 64, 0, stream>>>(norm, 8);
        scatter_xA<<<Ecnt / 256, 256, 0, stream>>>(A_vals, A_rows, A_cols, resid, v);
        vnorm_kernel<<<dim3(64, 8), 256, 0, stream>>>(u, v, norm);

        // r = z - eta * clip(v)
        rstep_kernel<<<512, 256, 0, stream>>>(zin, v, norm, etas + j, r);

        // CNN prox: WMMA convs + fused z-update
        conv1_wmma<<<1024, 256, 0, stream>>>(r, W1 + j * 288, act1);
        conv2_wmma<<<1024, 256, 0, stream>>>(act1, W2 + j * 9216, act2);
        conv3_zupdate<<<512, 256, 0, stream>>>(act2, W3 + j * 288, r, zout);
    }
}
